// LinearNF4_75797582840300
// MI455X (gfx1250) — matrix operand order repaired and checked by reading, not compiled, not run on hardware
//
#include <hip/hip_runtime.h>
#include <stdint.h>

typedef __attribute__((ext_vector_type(16))) _Float16 v16h;
typedef __attribute__((ext_vector_type(8)))  _Float16 v8h;
typedef __attribute__((ext_vector_type(4)))  _Float16 v4h;
typedef __attribute__((ext_vector_type(2)))  _Float16 h2;
typedef __attribute__((ext_vector_type(8)))  float    v8f;
typedef __attribute__((ext_vector_type(4)))  float    v4f;
typedef __attribute__((ext_vector_type(4)))  unsigned int u32x4;
typedef __attribute__((ext_vector_type(2)))  unsigned int u32x2;
typedef __attribute__((ext_vector_type(8)))  int      i32x8;
typedef __attribute__((ext_vector_type(4)))  int      i32x4;

#define K_DIM   4096
#define N_DIM   11008
#define M_DIM   512
#define NBLK    (N_DIM / 64)      // 172 scale blocks per k-row
#define KT      32                // k per step
#define NSTEPS  (K_DIM / KT)      // 128
#define TILE_M  128
#define TILE_N  128
#define A_STRIDE 40               // padded halves per A row
#define W_STRIDE 40               // padded halves per W^T row

#if __has_builtin(__builtin_amdgcn_tensor_load_to_lds) && __has_builtin(__builtin_amdgcn_s_wait_tensorcnt)
#define HAVE_TDM 1
#else
#define HAVE_TDM 0
#endif

__device__ __constant__ float NF4_TBL[16] = {
    -1.0f, -0.6961928009986877f, -0.5250730514526367f, -0.39491748809814453f,
    -0.28444138169288635f, -0.18477343022823334f, -0.09105003625154495f, 0.0f,
    0.07958029955625534f, 0.16093020141124725f, 0.24611230194568634f,
    0.33791524171829224f, 0.44070982933044434f, 0.5626170039176941f,
    0.7229568362236023f, 1.0f};

union V16 { v16h v; v8h h2[2]; };

#if HAVE_TDM
// Issue a TDM 2D tile load: 32 rows x 64 bytes, row stride 5504 B, into LDS.
__device__ __forceinline__ void tdm_load_packed(const unsigned char* g, unsigned int lds_off)
{
    const uint64_t ga = (uint64_t)(uintptr_t)g;
    u32x4 g0;
    g0[0] = 1u;                                            // count=1, user mode
    g0[1] = lds_off;                                       // lds_addr (bytes)
    g0[2] = (unsigned int)ga;                              // global_addr[31:0]
    g0[3] = (unsigned int)((ga >> 32) & 0x01FFFFFFu) | (2u << 30); // addr[56:32] | type=2
    i32x8 g1;
    g1[0] = 0;                       // wg_mask=0, data_size=0 (1B), no flags
    g1[1] = (int)(5504u << 16);      // tensor_dim0[15:0]  = 5504 (bytes/row)
    g1[2] = (int)(4096u << 16);      // tensor_dim1[15:0]  = 4096 (rows)
    g1[3] = (int)(64u << 16);        // tile_dim0 = 64 bytes
    g1[4] = 32;                      // tile_dim1 = 32 rows, tile_dim2 = 0
    g1[5] = 5504;                    // tensor_dim0_stride low32
    g1[6] = 0;                       // stride hi, dim1_stride lo
    g1[7] = 0;
    const i32x4 z4 = (i32x4){0, 0, 0, 0};
#if __clang_major__ >= 23
    const i32x8 z8 = (i32x8){0, 0, 0, 0, 0, 0, 0, 0};
    __builtin_amdgcn_tensor_load_to_lds(g0, g1, z4, z4, z8, 0);
#else
    __builtin_amdgcn_tensor_load_to_lds(g0, g1, z4, z4, 0);
#endif
}
#endif

__global__ __launch_bounds__(256) void nf4_wmma_gemm(
    const float* __restrict__ x,
    const unsigned char* __restrict__ packed,
    const float* __restrict__ scales,
    const float* __restrict__ bias,
    float* __restrict__ out)
{
    __shared__ _Float16 Asm[2][TILE_M * A_STRIDE];  // A tiles, f16 [m][k]
    __shared__ _Float16 Wsm[2][TILE_N * W_STRIDE];  // W^T tiles, f16 [n][k]
    __shared__ h2       Tbl2[256];                  // byte -> (f16 hi-nib, f16 lo-nib)
#if HAVE_TDM
    __shared__ __align__(16) unsigned char Pk[2][KT * 64];  // raw packed tile staging
#endif

    const int tid  = threadIdx.x;
    const int lane = tid & 31;
    const int wv   = tid >> 5;        // wave id 0..7
    const int lq   = lane & 15;
    const int lh   = lane >> 4;
    const int wrow = wv >> 1;         // 0..3 : 32-row strip
    const int wcol = wv & 1;          // 0..1 : 64-col strip

    const int nt = blockIdx.x * TILE_N;
    const int mt = blockIdx.y * TILE_M;

    // 256-entry pair table: x = value of HIGH nibble (even n), y = LOW nibble
    Tbl2[tid] = (h2){ (_Float16)NF4_TBL[tid >> 4], (_Float16)NF4_TBL[tid & 15] };

    v8f acc[8];
#pragma unroll
    for (int j = 0; j < 8; ++j) acc[j] = (v8f){0.f,0.f,0.f,0.f,0.f,0.f,0.f,0.f};

    // producer mapping (uniform across steps)
    const int am  = tid >> 1;            // A row 0..127
    const int akc = (tid & 1) * 16;      // A k-chunk {0,16}
    const int wk  = tid >> 3;            // W k-row 0..31
    const int wch = tid & 7;             // W 8-byte chunk 0..7
    const int wnl = wch * 16;            // local n base (16 n per thread)

#if HAVE_TDM
    typedef __attribute__((address_space(3))) unsigned char lds_uchar;
    const unsigned int pk_off[2] = {
        (unsigned int)(uintptr_t)(lds_uchar*)&Pk[0][0],
        (unsigned int)(uintptr_t)(lds_uchar*)&Pk[1][0] };
#endif

    // ---------------- staged producer registers ----------------
    v4f f0, f1, f2, f3;                  // A stage: 16 fp32
    uint32_t wb0, wb1;                   // W stage: 8 packed bytes
    float    wsc;                        // W stage: block scale

    // ---------------- prologue: fill buffer 0 for kt = 0 ----------------
#if HAVE_TDM
    if (wv == 0) tdm_load_packed(packed + (size_t)(nt >> 1), pk_off[0]);
#endif
    {
        const float* src = x + (size_t)(mt + am) * K_DIM + akc;
        f0 = *(const v4f*)(src + 0);  f1 = *(const v4f*)(src + 4);
        f2 = *(const v4f*)(src + 8);  f3 = *(const v4f*)(src + 12);
        wsc = scales[(size_t)wk * NBLK + (nt >> 6) + (wch >> 2)];
#if !HAVE_TDM
        const uint32_t* pk = (const uint32_t*)(packed + (size_t)wk * (N_DIM/2) + (nt >> 1) + wch * 8);
        wb0 = pk[0]; wb1 = pk[1];
#endif
    }
#if HAVE_TDM
    if (wv == 0) __builtin_amdgcn_s_wait_tensorcnt(0);
#endif
    __syncthreads();   // Tbl2 ready (+ TDM tile visible)
    {
#if HAVE_TDM
        const u32x2 wb = *(const u32x2*)&Pk[0][wk * 64 + wch * 8];
        wb0 = wb[0]; wb1 = wb[1];
#endif
        _Float16* dst = &Asm[0][am * A_STRIDE + akc];
        *(v4h*)(dst + 0)  = __builtin_convertvector(f0, v4h);
        *(v4h*)(dst + 4)  = __builtin_convertvector(f1, v4h);
        *(v4h*)(dst + 8)  = __builtin_convertvector(f2, v4h);
        *(v4h*)(dst + 12) = __builtin_convertvector(f3, v4h);
        const _Float16 hs = (_Float16)wsc;
        const h2 s2 = (h2){hs, hs};
#pragma unroll
        for (int i = 0; i < 8; ++i) {
            const unsigned code = ((i < 4 ? wb0 : wb1) >> ((i & 3) * 8)) & 0xFF;
            const h2 val = Tbl2[code] * s2;
            Wsm[0][(wnl + 2*i + 0) * W_STRIDE + wk] = val.x;
            Wsm[0][(wnl + 2*i + 1) * W_STRIDE + wk] = val.y;
        }
    }
    __syncthreads();

    // ---------------- main pipelined K loop ----------------
    for (int s = 0; s < NSTEPS; ++s) {
        const int  cur  = s & 1;
        const int  nxt  = cur ^ 1;
        const bool more = (s + 1) < NSTEPS;

        // issue next tile's data movement early (hidden behind WMMA)
        if (more) {
            const int kt = (s + 1) * KT;
#if HAVE_TDM
            if (wv == 0)
                tdm_load_packed(packed + (size_t)kt * (N_DIM/2) + (nt >> 1), pk_off[nxt]);
#else
            const uint32_t* pk = (const uint32_t*)(packed + (size_t)(kt + wk) * (N_DIM/2) + (nt >> 1) + wch * 8);
            wb0 = pk[0]; wb1 = pk[1];
#endif
            const float* src = x + (size_t)(mt + am) * K_DIM + kt + akc;
            f0 = *(const v4f*)(src + 0);  f1 = *(const v4f*)(src + 4);
            f2 = *(const v4f*)(src + 8);  f3 = *(const v4f*)(src + 12);
            wsc = scales[(size_t)(kt + wk) * NBLK + (nt >> 6) + (wch >> 2)];
            if (s + 2 < NSTEPS)
                __builtin_prefetch(scales + (size_t)(kt + KT + wk) * NBLK + (nt >> 6), 0, 1);
        }

        // ---- compute from buffer cur: batch fragment loads, then 8 WMMAs ----
        {
            const _Float16* Ac = Asm[cur];
            const _Float16* Wc = Wsm[cur];
            V16 a[2], b[4];
            const int kOff = lh * 8;
            const int kb   = lh * 8;
#pragma unroll
            for (int mi = 0; mi < 2; ++mi) {
                const _Float16* ap = &Ac[(wrow * 32 + mi * 16 + lq) * A_STRIDE + kOff];
                a[mi].h2[0] = *(const v8h*)(ap + 0);   // K = kOff..kOff+7
                a[mi].h2[1] = *(const v8h*)(ap + 16);  // K = kOff+16..kOff+23
            }
#pragma unroll
            for (int ni = 0; ni < 4; ++ni) {
                const _Float16* bp = &Wc[(wcol * 64 + ni * 16 + lq) * W_STRIDE + kb];
                b[ni].h2[0] = *(const v8h*)(bp + 0);   // K = kb..kb+7
                b[ni].h2[1] = *(const v8h*)(bp + 16);  // K = kb+16..kb+23
            }
#pragma unroll
            for (int mi = 0; mi < 2; ++mi)
#pragma unroll
                for (int ni = 0; ni < 4; ++ni)
                    acc[mi * 4 + ni] = __builtin_amdgcn_wmma_f32_16x16x32_f16(
                        false, a[mi].v, false, b[ni].v, (short)0,
                        acc[mi * 4 + ni], false, false);
        }

        // ---- dequant/convert next tile into the other buffer ----
        if (more) {
#if HAVE_TDM
            if (wv == 0) __builtin_amdgcn_s_wait_tensorcnt(0);
            __syncthreads();   // TDM tile visible to all waves
            const u32x2 wb = *(const u32x2*)&Pk[nxt][wk * 64 + wch * 8];
            wb0 = wb[0]; wb1 = wb[1];
#endif
            _Float16* dst = &Asm[nxt][am * A_STRIDE + akc];
            *(v4h*)(dst + 0)  = __builtin_convertvector(f0, v4h);
            *(v4h*)(dst + 4)  = __builtin_convertvector(f1, v4h);
            *(v4h*)(dst + 8)  = __builtin_convertvector(f2, v4h);
            *(v4h*)(dst + 12) = __builtin_convertvector(f3, v4h);
            const _Float16 hs = (_Float16)wsc;
            const h2 s2 = (h2){hs, hs};
#pragma unroll
            for (int i = 0; i < 8; ++i) {
                const unsigned code = ((i < 4 ? wb0 : wb1) >> ((i & 3) * 8)) & 0xFF;
                const h2 val = Tbl2[code] * s2;
                Wsm[nxt][(wnl + 2*i + 0) * W_STRIDE + wk] = val.x;
                Wsm[nxt][(wnl + 2*i + 1) * W_STRIDE + wk] = val.y;
            }
        }
        __syncthreads();
    }

    // ---- epilogue: C layout -> global, add bias ----
#pragma unroll
    for (int mi = 0; mi < 2; ++mi) {
#pragma unroll
        for (int ni = 0; ni < 4; ++ni) {
            const int col = nt + wcol * 64 + ni * 16 + lq;
            const float bv = bias[col];
            const v8f c = acc[mi * 4 + ni];
#pragma unroll
            for (int r = 0; r < 8; ++r) {
                const int row = mt + wrow * 32 + mi * 16 + r + 8 * lh;
                out[(size_t)row * N_DIM + col] = c[r] + bv;
            }
        }
    }
}

extern "C" void kernel_launch(void* const* d_in, const int* in_sizes, int n_in,
                              void* d_out, int out_size, void* d_ws, size_t ws_size,
                              hipStream_t stream) {
    (void)in_sizes; (void)n_in; (void)d_ws; (void)ws_size; (void)out_size;
    const float*         x      = (const float*)d_in[0];
    const unsigned char* packed = (const unsigned char*)d_in[1];
    const float*         scales = (const float*)d_in[2];
    const float*         bias   = (const float*)d_in[3];
    float*               out    = (float*)d_out;

    dim3 grid(N_DIM / TILE_N, M_DIM / TILE_M);  // (86, 4)
    dim3 block(256);
    nf4_wmma_gemm<<<grid, block, 0, stream>>>(x, packed, scales, bias, out);
}
